// Block_75823352644327
// MI455X (gfx1250) — compile-verified, matrix operand order checked by reading, not run on hardware
//
#include <hip/hip_runtime.h>
#include <cstdint>
#include <cstddef>

// ---------------------------------------------------------------------------
// Types for CDNA5 WMMA (wave32): v16bf A/B operands, v8f f32 accumulator
// ---------------------------------------------------------------------------
typedef __bf16 bf16_t;
typedef __attribute__((ext_vector_type(16))) __bf16 v16bf;
typedef __attribute__((ext_vector_type(8)))  __bf16 v8bf;
typedef __attribute__((ext_vector_type(8)))  float  v8f;

#define BB   8
#define SS   1024
#define DD   1024
#define HH   16
#define HDIM 64
#define FFN_ 4096
#define MM   (BB * SS)

__device__ __forceinline__ bf16_t f2bf(float f) {
  uint32_t u = __builtin_bit_cast(uint32_t, f);
  uint32_t r = u + 0x7FFFu + ((u >> 16) & 1u);   // round-to-nearest-even
  uint16_t h = (uint16_t)(r >> 16);
  return __builtin_bit_cast(bf16_t, h);
}

// Build a v16bf operand fragment from two contiguous 8-element halves.
// Layout per CDNA5 ISA (16-bit operand 16x32): lanes 0-15 K={0..7,16..23},
// lanes 16-31 K={8..15,24..31} -> caller passes p (K base) and p+16.
__device__ __forceinline__ v16bf frag16(const bf16_t* p0, const bf16_t* p1) {
  v8bf lo = *(const v8bf*)p0;
  v8bf hi = *(const v8bf*)p1;
  v16bf r;
#pragma unroll
  for (int j = 0; j < 8; ++j) { r[j] = lo[j]; r[j + 8] = hi[j]; }
  return r;
}

__device__ __forceinline__ v8f wmma_bf16(v16bf a, v16bf b, v8f c) {
  return __builtin_amdgcn_wmma_f32_16x16x32_bf16(false, a, false, b,
                                                 (short)0, c, false, false);
}

// Async global->LDS copy of 16 bytes (GLOBAL_LOAD_ASYNC_TO_LDS_B128,
// tracked by ASYNCcnt). dst_lds is an LDS byte offset, src a global pointer.
__device__ __forceinline__ void async_copy_b128(const void* src, void* dst_lds) {
  unsigned dst = (unsigned)(size_t)dst_lds;  // low 32 bits = LDS offset
  asm volatile("global_load_async_to_lds_b128 %0, %1, off"
               :: "v"(dst), "v"(src) : "memory");
}
__device__ __forceinline__ void wait_async0() {
  asm volatile("s_wait_asynccnt 0x0" ::: "memory");
}

// ---------------------------------------------------------------------------
// Weight cast + transpose: out[n*K + k] = bf16(in[k*N + n]). 32x32 LDS tile.
// ---------------------------------------------------------------------------
__global__ __launch_bounds__(256) void castT_kernel(
    const float* __restrict__ in, bf16_t* __restrict__ out, int Kd, int Nd) {
  __shared__ float tile[32][33];
  const int nb = blockIdx.x * 32, kb = blockIdx.y * 32;
  const int tx = threadIdx.x & 31, ty = threadIdx.x >> 5;  // ty 0..7
#pragma unroll
  for (int i = 0; i < 32; i += 8)
    tile[ty + i][tx] = in[(long)(kb + ty + i) * Nd + nb + tx];
  __syncthreads();
#pragma unroll
  for (int i = 0; i < 32; i += 8)
    out[(long)(nb + ty + i) * Kd + kb + tx] = f2bf(tile[tx][ty + i]);
}

// ---------------------------------------------------------------------------
// LayerNorm over D=1024, fp32 in -> bf16 out. One block (256 thr) per row.
// ---------------------------------------------------------------------------
__global__ __launch_bounds__(256) void ln_bf16_kernel(
    const float* __restrict__ X, const float* __restrict__ g,
    const float* __restrict__ b, bf16_t* __restrict__ out) {
  __shared__ float red[2][8];
  const int row = blockIdx.x, t = threadIdx.x, lane = t & 31, w = t >> 5;
  const float* xr = X + (long)row * DD;
  float4 v = *(const float4*)&xr[t * 4];
  float s  = v.x + v.y + v.z + v.w;
  float ss = v.x * v.x + v.y * v.y + v.z * v.z + v.w * v.w;
#pragma unroll
  for (int o = 16; o >= 1; o >>= 1) {
    s  += __shfl_xor(s,  o, 32);
    ss += __shfl_xor(ss, o, 32);
  }
  if (lane == 0) { red[0][w] = s; red[1][w] = ss; }
  __syncthreads();
  float S = 0.f, S2 = 0.f;
#pragma unroll
  for (int i = 0; i < 8; ++i) { S += red[0][i]; S2 += red[1][i]; }
  const float mean = S * (1.0f / DD);
  const float var  = S2 * (1.0f / DD) - mean * mean;
  const float rstd = rsqrtf(var + 1e-5f);
  bf16_t* orow = out + (long)row * DD;
  const float xs[4] = {v.x, v.y, v.z, v.w};
#pragma unroll
  for (int j = 0; j < 4; ++j) {
    int c = t * 4 + j;
    orow[c] = f2bf((xs[j] - mean) * rstd * g[c] + b[c]);
  }
}

// ---------------------------------------------------------------------------
// Tiled bf16 GEMM, f32 accumulate via v_wmma_f32_16x16x32_bf16.
//   C[M,N] = epilogue( A[M,K] @ BT[N,K]^T )   (BT = pre-transposed weights)
// Tile 128x128, K-step 32, double-buffered LDS filled with async global->LDS
// DMA (ASYNCcnt). 256 threads = 8 waves; wave (wm,wn) owns 32x64 = 2x4 WMMAs.
// Epilogue: +bias, act (0/1=GELU/2=ReLU), *scale, +addC, +addR,
// store_mode: 0 f32, 1 bf16 row-major, 2 bf16 per-head transposed [b,h,d,s].
// ---------------------------------------------------------------------------
#define TKS 32
__global__ __launch_bounds__(256) void gemm_bf16_kernel(
    const bf16_t* __restrict__ A, const bf16_t* __restrict__ BT,
    int Mi, int Ni, int Ki,
    const float* __restrict__ bias,
    float* __restrict__ Cf, bf16_t* __restrict__ Cb,
    const float* __restrict__ addC, const float* __restrict__ addR,
    float scale, int act, int store_mode) {
  __shared__ __align__(16) bf16_t As[2][128][TKS + 8];  // 80B pitch
  __shared__ __align__(16) bf16_t Bs[2][128][TKS + 8];  // [n][k]

  const int t = threadIdx.x, lane = t & 31, w = t >> 5;
  const int m0 = blockIdx.y * 128, n0 = blockIdx.x * 128;
  const int wm = w & 3, wn = w >> 2;  // 4 x 2 wave grid

  v8f acc[2][4] = {};

  auto stage = [&](int buf, int kk) {
#pragma unroll
    for (int i = 0; i < 2; ++i) {  // A tile: 512 chunks of 8 bf16
      int c = t + i * 256;
      int r = c >> 2, kc = (c & 3) * 8;
      async_copy_b128(&A[(long)(m0 + r) * Ki + kk + kc], &As[buf][r][kc]);
    }
#pragma unroll
    for (int i = 0; i < 2; ++i) {  // B tile (same geometry; BT row-major)
      int c = t + i * 256;
      int r = c >> 2, kc = (c & 3) * 8;
      if (n0 + r < Ni)  // OOB rows feed only discarded output tiles
        async_copy_b128(&BT[(long)(n0 + r) * Ki + kk + kc], &Bs[buf][r][kc]);
    }
  };

  stage(0, 0);
  wait_async0();
  __syncthreads();

  int cur = 0;
  for (int kk = 0; kk < Ki; kk += TKS) {
    if (kk + TKS < Ki) stage(cur ^ 1, kk + TKS);  // prefetch next K-step

    v16bf af[2], bfr[4];
#pragma unroll
    for (int i = 0; i < 2; ++i) {
      int r = wm * 32 + i * 16 + (lane & 15);
      const bf16_t* p = &As[cur][r][(lane >> 4) * 8];
      af[i] = frag16(p, p + 16);
    }
#pragma unroll
    for (int j = 0; j < 4; ++j) {
      int r = wn * 64 + j * 16 + (lane & 15);
      const bf16_t* p = &Bs[cur][r][(lane >> 4) * 8];
      bfr[j] = frag16(p, p + 16);
    }
#pragma unroll
    for (int i = 0; i < 2; ++i)
#pragma unroll
      for (int j = 0; j < 4; ++j)
        acc[i][j] = wmma_bf16(af[i], bfr[j], acc[i][j]);

    wait_async0();     // next buffer's DMA complete (this wave)
    __syncthreads();   // ... and all waves done reading/writing
    cur ^= 1;
  }

  // ---- epilogue; D layout: VGPR e -> M = 8*(lane>>4)+e, N = lane&15 ----
#pragma unroll
  for (int i = 0; i < 2; ++i)
#pragma unroll
    for (int j = 0; j < 4; ++j)
#pragma unroll
      for (int e = 0; e < 8; ++e) {
        int row = m0 + wm * 32 + i * 16 + ((lane >> 4) * 8) + e;
        int col = n0 + wn * 64 + j * 16 + (lane & 15);
        if (row < Mi && col < Ni) {
          float v = acc[i][j][e];
          if (bias) v += bias[col];
          if (act == 1) v = 0.5f * v * (1.f + erff(v * 0.70710678118f));
          else if (act == 2) v = fmaxf(v, 0.f);
          v *= scale;
          long idx = (long)row * Ni + col;
          if (addC) v += addC[idx];
          if (addR) v += addR[idx];
          if (store_mode == 0) Cf[idx] = v;
          else if (store_mode == 1) Cb[idx] = f2bf(v);
          else {  // per-head transposed: VT[b, h, d, s]
            long idx2 = ((long)((row >> 10) * HH + (col >> 6)) * HDIM +
                         (col & 63)) * SS + (row & 1023);
            Cb[idx2] = f2bf(v);
          }
        }
      }
}

// ---------------------------------------------------------------------------
// Attention: one WG per (b, h, 32-query block). Score rows (32x1024 fp32,
// 128 KB) + bf16 probs (64 KB) in the 320 KB CDNA5 LDS (dynamic). Q/K/VT
// fragments load straight from global: contraction dims are contiguous and
// match the WMMA operand layout, so no LDS staging/transposes at all.
// ---------------------------------------------------------------------------
__global__ __launch_bounds__(256) void attn_kernel(
    const bf16_t* __restrict__ Q, const bf16_t* __restrict__ K,
    const bf16_t* __restrict__ VT,  // [B, H, HDIM, S]
    bf16_t* __restrict__ O) {
  extern __shared__ char smem[];
  float*  sc = (float*)smem;                   // [32][1024] f32
  bf16_t* pb = (bf16_t*)(smem + 32 * SS * 4);  // [32][1024] bf16

  const int t = threadIdx.x, lane = t & 31, w = t >> 5;
  const int qb0 = blockIdx.x * 32;
  const int bb = blockIdx.y / HH, hh = blockIdx.y % HH;
  const long base = (long)(bb * SS) * DD + hh * HDIM;

  // ---- preload Q fragments: 2 q-tiles x 2 K-steps (d 0..31, 32..63) ----
  v16bf qa[2][2];
#pragma unroll
  for (int qt = 0; qt < 2; ++qt)
#pragma unroll
    for (int ds = 0; ds < 2; ++ds) {
      int r = qb0 + qt * 16 + (lane & 15);
      int c = ds * 32 + ((lane >> 4) * 8);
      const bf16_t* p = Q + base + (long)r * DD + c;
      qa[qt][ds] = frag16(p, p + 16);
    }

  // ---- phase 1: S = (Q K^T) * HD^-0.5 ; wave w owns key columns w*128.. ----
  const int kc0 = w * 128;
  for (int kt = 0; kt < 8; ++kt) {
    int scol = kc0 + kt * 16;
    v16bf kb[2];
#pragma unroll
    for (int ds = 0; ds < 2; ++ds) {
      int r = scol + (lane & 15);
      int c = ds * 32 + ((lane >> 4) * 8);
      const bf16_t* p = K + base + (long)r * DD + c;
      kb[ds] = frag16(p, p + 16);
    }
#pragma unroll
    for (int qt = 0; qt < 2; ++qt) {
      v8f acc = {};
      acc = wmma_bf16(qa[qt][0], kb[0], acc);
      acc = wmma_bf16(qa[qt][1], kb[1], acc);
#pragma unroll
      for (int e = 0; e < 8; ++e) {
        int q = qt * 16 + ((lane >> 4) * 8) + e;
        sc[q * SS + scol + (lane & 15)] = acc[e] * 0.125f;  // HD^-0.5
      }
    }
  }
  __syncthreads();

  // ---- phase 2: softmax; 8 lanes per row, shfl_xor reduction (wave32) ----
  {
    const int row = t >> 3, part = t & 7;
    float* srow = sc + row * SS + part * 128;
    float m = -3.0e38f;
    for (int c = 0; c < 128; ++c) m = fmaxf(m, srow[c]);
    m = fmaxf(m, __shfl_xor(m, 1, 32));
    m = fmaxf(m, __shfl_xor(m, 2, 32));
    m = fmaxf(m, __shfl_xor(m, 4, 32));
    float s = 0.f;
    for (int c = 0; c < 128; ++c) s += __expf(srow[c] - m);
    s += __shfl_xor(s, 1, 32);
    s += __shfl_xor(s, 2, 32);
    s += __shfl_xor(s, 4, 32);
    const float inv = 1.f / s;
    bf16_t* prow = pb + row * SS + part * 128;
    for (int c = 0; c < 128; ++c) prow[c] = f2bf(__expf(srow[c] - m) * inv);
  }
  __syncthreads();

  // ---- phase 3: O = P @ V ; V fragments direct from global VT[b,h,d,s] ----
  const int qt = w & 1, dt = w >> 1;  // 2 x 4 wave grid
  const bf16_t* vbase = VT + (long)(bb * HH + hh) * HDIM * SS;
  v8f acc = {};
  for (int ks = 0; ks < SS; ks += 32) {
    const bf16_t* ap =
        pb + (qt * 16 + (lane & 15)) * SS + ks + ((lane >> 4) * 8);
    v16bf af = frag16(ap, ap + 16);
    const bf16_t* bp =
        vbase + (long)(dt * 16 + (lane & 15)) * SS + ks + ((lane >> 4) * 8);
    v16bf bfg = frag16(bp, bp + 16);
    acc = wmma_bf16(af, bfg, acc);
  }
#pragma unroll
  for (int e = 0; e < 8; ++e) {
    int q = qb0 + qt * 16 + ((lane >> 4) * 8) + e;
    int c = dt * 16 + (lane & 15);
    O[(long)(bb * SS + q) * DD + hh * HDIM + c] = f2bf(acc[e]);
  }
}

// ---------------------------------------------------------------------------
// Host-side orchestration
// ---------------------------------------------------------------------------
extern "C" void kernel_launch(void* const* d_in, const int* in_sizes, int n_in,
                              void* d_out, int out_size, void* d_ws,
                              size_t ws_size, hipStream_t stream) {
  (void)in_sizes; (void)n_in; (void)out_size; (void)ws_size;
  const float* x     = (const float*)d_in[0];
  const float* ln1_g = (const float*)d_in[1];
  const float* ln1_b = (const float*)d_in[2];
  const float* wq    = (const float*)d_in[3];
  const float* wk    = (const float*)d_in[4];
  const float* wv    = (const float*)d_in[5];
  const float* wo    = (const float*)d_in[6];
  const float* bo    = (const float*)d_in[7];
  const float* ln2_g = (const float*)d_in[8];
  const float* ln2_b = (const float*)d_in[9];
  const float* w1    = (const float*)d_in[10];
  const float* b1    = (const float*)d_in[11];
  const float* w2    = (const float*)d_in[12];
  const float* b2    = (const float*)d_in[13];
  const float* aln_g = (const float*)d_in[14];
  const float* aln_b = (const float*)d_in[15];
  const float* wd    = (const float*)d_in[16];
  const float* bd    = (const float*)d_in[17];
  const float* wu    = (const float*)d_in[18];
  const float* bu    = (const float*)d_in[19];
  float* out = (float*)d_out;

  char* ws = (char*)d_ws;
  size_t off = 0;
  auto carve = [&](size_t bytes) {
    void* p = ws + off;
    off += (bytes + 255) & ~(size_t)255;
    return p;
  };
  // transposed bf16 weights: WT[n][k]
  bf16_t* wqT = (bf16_t*)carve((size_t)DD * DD * 2);
  bf16_t* wkT = (bf16_t*)carve((size_t)DD * DD * 2);
  bf16_t* wvT = (bf16_t*)carve((size_t)DD * DD * 2);
  bf16_t* woT = (bf16_t*)carve((size_t)DD * DD * 2);
  bf16_t* w1T = (bf16_t*)carve((size_t)DD * FFN_ * 2);
  bf16_t* w2T = (bf16_t*)carve((size_t)FFN_ * DD * 2);
  bf16_t* wdT = (bf16_t*)carve((size_t)DD * 64 * 2);
  bf16_t* wuT = (bf16_t*)carve((size_t)64 * DD * 2);
  bf16_t* hb  = (bf16_t*)carve((size_t)MM * DD * 2);  // LN1 / adapter-LN out
  bf16_t* qb  = (bf16_t*)carve((size_t)MM * DD * 2);  // Q; reused as LN2 out
  bf16_t* kb  = (bf16_t*)carve((size_t)MM * DD * 2);
  bf16_t* vtb = (bf16_t*)carve((size_t)MM * DD * 2);  // V, [B,H,HDIM,S]
  bf16_t* atb = (bf16_t*)carve((size_t)MM * DD * 2);  // attention output
  float*  x1f = (float*)carve((size_t)MM * DD * 4);   // residual after attn
  bf16_t* adb = (bf16_t*)carve((size_t)MM * 64 * 2);  // relu bottleneck
  bf16_t* g1b = (bf16_t*)carve((size_t)MM * FFN_ * 2);

  dim3 blk(256);
  // --- cast+transpose weights fp32[K,N] -> bf16[N,K] ---
  castT_kernel<<<dim3(DD / 32, DD / 32), blk, 0, stream>>>(wq, wqT, DD, DD);
  castT_kernel<<<dim3(DD / 32, DD / 32), blk, 0, stream>>>(wk, wkT, DD, DD);
  castT_kernel<<<dim3(DD / 32, DD / 32), blk, 0, stream>>>(wv, wvT, DD, DD);
  castT_kernel<<<dim3(DD / 32, DD / 32), blk, 0, stream>>>(wo, woT, DD, DD);
  castT_kernel<<<dim3(FFN_ / 32, DD / 32), blk, 0, stream>>>(w1, w1T, DD, FFN_);
  castT_kernel<<<dim3(DD / 32, FFN_ / 32), blk, 0, stream>>>(w2, w2T, FFN_, DD);
  castT_kernel<<<dim3(64 / 32, DD / 32), blk, 0, stream>>>(wd, wdT, DD, 64);
  castT_kernel<<<dim3(DD / 32, 64 / 32), blk, 0, stream>>>(wu, wuT, 64, DD);

  dim3 gD((DD + 127) / 128, MM / 128);    // N=1024
  dim3 gF((FFN_ + 127) / 128, MM / 128);  // N=4096
  dim3 gA(1, MM / 128);                   // N=64

  // --- h = LN1(x) ---
  ln_bf16_kernel<<<MM, blk, 0, stream>>>(x, ln1_g, ln1_b, hb);
  // --- Q,K,V projections (V stored per-head transposed) ---
  gemm_bf16_kernel<<<gD, blk, 0, stream>>>(hb, wqT, MM, DD, DD, nullptr,
      nullptr, qb, nullptr, nullptr, 1.f, 0, 1);
  gemm_bf16_kernel<<<gD, blk, 0, stream>>>(hb, wkT, MM, DD, DD, nullptr,
      nullptr, kb, nullptr, nullptr, 1.f, 0, 1);
  gemm_bf16_kernel<<<gD, blk, 0, stream>>>(hb, wvT, MM, DD, DD, nullptr,
      nullptr, vtb, nullptr, nullptr, 1.f, 0, 2);
  // --- attention ---
  size_t attn_lds = (size_t)32 * SS * 4 + (size_t)32 * SS * 2;  // 192 KB
  attn_kernel<<<dim3(SS / 32, BB * HH), blk, attn_lds, stream>>>(qb, kb, vtb, atb);
  // --- x1 = x + attn @ wo + bo ---
  gemm_bf16_kernel<<<gD, blk, 0, stream>>>(atb, woT, MM, DD, DD, bo,
      x1f, nullptr, nullptr, x, 1.f, 0, 0);
  // --- adapter: a = LN(x1); ad = relu(a@wd+bd); out = (ad@wu+bu)*0.1 ---
  ln_bf16_kernel<<<MM, blk, 0, stream>>>(x1f, aln_g, aln_b, hb);
  gemm_bf16_kernel<<<gA, blk, 0, stream>>>(hb, wdT, MM, 64, DD, bd,
      nullptr, adb, nullptr, nullptr, 1.f, 2, 1);
  gemm_bf16_kernel<<<gD, blk, 0, stream>>>(adb, wuT, MM, DD, 64, bu,
      out, nullptr, nullptr, nullptr, 0.1f, 0, 0);
  // --- MLP: h2 = LN2(x1); g1 = gelu(h2@w1+b1); out += x1 + g1@w2 + b2 ---
  ln_bf16_kernel<<<MM, blk, 0, stream>>>(x1f, ln2_g, ln2_b, qb);
  gemm_bf16_kernel<<<gF, blk, 0, stream>>>(qb, w1T, MM, FFN_, DD, b1,
      nullptr, g1b, nullptr, nullptr, 1.f, 1, 1);
  gemm_bf16_kernel<<<gD, blk, 0, stream>>>(g1b, w2T, MM, DD, FFN_, b2,
      out, nullptr, out, x1f, 1.f, 0, 0);
}
